// HeteroGNN_10651518894758
// MI455X (gfx1250) — compile-verified, matrix operand order checked by reading, not run on hardware
//
#include <hip/hip_runtime.h>
#include <cstdint>
#include <cstddef>

// ---------------------------------------------------------------------------
// HeteroGNN (3-layer hetero SAGEConv) for gfx1250.
//  - scatter/segment-mean with native f32 global atomics (coalesced, lane=feature)
//  - GEMMs via v_wmma_f32_16x16x32_bf16 (f32 accumulate), 16x32 tile per wave
// ---------------------------------------------------------------------------

#define HDIM 256          // hidden size (fixed by the reference)
#define NLAYER 3

typedef __bf16 bf16_t;
typedef __attribute__((ext_vector_type(16))) __bf16 v16bf;
typedef __attribute__((ext_vector_type(8)))  __bf16 v8bf;
typedef __attribute__((ext_vector_type(8)))  float  v8f;

__device__ __forceinline__ bf16_t f32_to_bf16(float f) {
  unsigned u = __builtin_bit_cast(unsigned, f);
  u = (u + 0x7FFFu + ((u >> 16) & 1u)) >> 16;        // round-to-nearest-even
  unsigned short s = (unsigned short)u;
  return __builtin_bit_cast(bf16_t, s);
}

// ----------------------------- utility kernels -----------------------------

__global__ __launch_bounds__(256) void zero_f32(float* __restrict__ p, long long n) {
  long long i = (long long)blockIdx.x * 256 + threadIdx.x;
  if (i < n) p[i] = 0.0f;
}

__global__ __launch_bounds__(256) void cvt_f32_bf16(const float* __restrict__ x,
                                                    bf16_t* __restrict__ y, long long n) {
  long long i = (long long)blockIdx.x * 256 + threadIdx.x;
  if (i < n) y[i] = f32_to_bf16(x[i]);
}

// W[mat][k][j] (f32, row-major HxH) -> Wt[mat][j][k] (bf16)
__global__ __launch_bounds__(256) void transpose_cvt_w(const float* __restrict__ W,
                                                       bf16_t* __restrict__ Wt,
                                                       long long total) {
  long long i = (long long)blockIdx.x * 256 + threadIdx.x;
  if (i >= total) return;
  long long mat = i / (HDIM * HDIM);
  int rem = (int)(i - mat * (HDIM * HDIM));
  int k = rem >> 8;          // row in W
  int j = rem & (HDIM - 1);  // col in W
  Wt[mat * (HDIM * HDIM) + (long long)j * HDIM + k] = f32_to_bf16(W[i]);
}

// One block per edge; lane t owns feature t. Coalesced read + f32 global atomic add.
__global__ __launch_bounds__(256) void scatter_add(const int* __restrict__ src,
                                                   const int* __restrict__ dst,
                                                   const float* __restrict__ x,
                                                   float* __restrict__ agg,
                                                   float* __restrict__ cnt) {
  int e = blockIdx.x;
  int s = src[e];
  int d = dst[e];
  float v = x[(size_t)s * HDIM + threadIdx.x];
  __hip_atomic_fetch_add(agg + (size_t)d * HDIM + threadIdx.x, v,
                         __ATOMIC_RELAXED, __HIP_MEMORY_SCOPE_AGENT);
  if (threadIdx.x == 0)
    __hip_atomic_fetch_add(cnt + d, 1.0f, __ATOMIC_RELAXED, __HIP_MEMORY_SCOPE_AGENT);
}

// mean = agg / max(cnt,1)  -> bf16
__global__ __launch_bounds__(256) void mean_cvt(const float* __restrict__ agg,
                                                const float* __restrict__ cnt,
                                                bf16_t* __restrict__ out, long long n) {
  long long i = (long long)blockIdx.x * 256 + threadIdx.x;
  if (i >= n) return;
  float c = cnt[i >> 8];
  out[i] = f32_to_bf16(agg[i] / fmaxf(c, 1.0f));
}

// ------------------------------- WMMA GEMM ---------------------------------
// out[M x 256] (+)= A0[M x 256] @ B0t^T + A1[M x 256] @ B1t^T + bias
// B*t are pre-transposed (Wt[j][k]) so B fragments are contiguous-row loads.
// One wave = one 16x32 output tile (two 16x16 WMMA accumulators): each A
// fragment is loaded once and fed to two WMMAs. 8 waves/block cover all 16
// N-tiles of one 16-row M-tile, so A rows stay hot in WGP$/L2.

union V16u { v16bf v; v8bf h[2]; };

template <int BETA>
__global__ __launch_bounds__(256) void gemm2_wmma(const bf16_t* __restrict__ A0,
                                                  const bf16_t* __restrict__ B0t,
                                                  const bf16_t* __restrict__ A1,
                                                  const bf16_t* __restrict__ B1t,
                                                  const float* __restrict__ bias,
                                                  float* __restrict__ out) {
  const int wave = threadIdx.x >> 5;
  const int lane = threadIdx.x & 31;
  const int half = lane >> 4;        // 0: lanes 0-15, 1: lanes 16-31
  const int l16  = lane & 15;
  const int tileM = blockIdx.x;

  const int row  = tileM * 16 + l16;   // A row this lane supplies
  const int col0 = wave * 32 + l16;    // first C col this lane supplies
  const int col1 = col0 + 16;          // second C col

  const bf16_t* a0p  = A0 + (size_t)row * HDIM;
  const bf16_t* a1p  = A1 + (size_t)row * HDIM;
  const bf16_t* b0p0 = B0t + (size_t)col0 * HDIM;
  const bf16_t* b0p1 = B0t + (size_t)col1 * HDIM;
  const bf16_t* b1p0 = B1t + (size_t)col0 * HDIM;
  const bf16_t* b1p1 = B1t + (size_t)col1 * HDIM;

  v8f acc0 = {};
  v8f acc1 = {};
#pragma unroll
  for (int k0 = 0; k0 < HDIM; k0 += 32) {
    const int klo = k0 + 8 * half;   // lanes 0-15: K 0..7 / 16..23 ; 16-31: +8
    V16u a, b0, b1;
    // pair 0: A0 @ B0t^T
    a.h[0]  = *(const v8bf*)(a0p + klo);
    a.h[1]  = *(const v8bf*)(a0p + klo + 16);
    b0.h[0] = *(const v8bf*)(b0p0 + klo);
    b0.h[1] = *(const v8bf*)(b0p0 + klo + 16);
    b1.h[0] = *(const v8bf*)(b0p1 + klo);
    b1.h[1] = *(const v8bf*)(b0p1 + klo + 16);
    acc0 = __builtin_amdgcn_wmma_f32_16x16x32_bf16(false, a.v, false, b0.v,
                                                   (short)0, acc0, false, false);
    acc1 = __builtin_amdgcn_wmma_f32_16x16x32_bf16(false, a.v, false, b1.v,
                                                   (short)0, acc1, false, false);
    // pair 1: A1 @ B1t^T
    a.h[0]  = *(const v8bf*)(a1p + klo);
    a.h[1]  = *(const v8bf*)(a1p + klo + 16);
    b0.h[0] = *(const v8bf*)(b1p0 + klo);
    b0.h[1] = *(const v8bf*)(b1p0 + klo + 16);
    b1.h[0] = *(const v8bf*)(b1p1 + klo);
    b1.h[1] = *(const v8bf*)(b1p1 + klo + 16);
    acc0 = __builtin_amdgcn_wmma_f32_16x16x32_bf16(false, a.v, false, b0.v,
                                                   (short)0, acc0, false, false);
    acc1 = __builtin_amdgcn_wmma_f32_16x16x32_bf16(false, a.v, false, b1.v,
                                                   (short)0, acc1, false, false);
  }

  const float bcol0 = bias[col0];
  const float bcol1 = bias[col1];
  const int m0 = tileM * 16 + 8 * half;  // C VGPR r -> M = r (+8 for upper lanes)
#pragma unroll
  for (int r = 0; r < 8; ++r) {
    size_t idx0 = (size_t)(m0 + r) * HDIM + col0;
    size_t idx1 = (size_t)(m0 + r) * HDIM + col1;
    float prev0 = BETA ? out[idx0] : 0.0f;
    float prev1 = BETA ? out[idx1] : 0.0f;
    out[idx0] = prev0 + acc0[r] + bcol0;
    out[idx1] = prev1 + acc1[r] + bcol1;
  }
}

// ------------------------------- host side ---------------------------------

static inline size_t align256(size_t x) { return (x + 255) & ~(size_t)255; }

extern "C" void kernel_launch(void* const* d_in, const int* in_sizes, int n_in,
                              void* d_out, int out_size, void* d_ws, size_t ws_size,
                              hipStream_t stream) {
  (void)n_in; (void)out_size; (void)ws_size;

  const float* x_author = (const float*)d_in[0];
  const float* x_paper  = (const float*)d_in[1];
  const int*   e_writes = (const int*)d_in[2];   // [2,E] author->paper
  const int*   e_rev    = (const int*)d_in[3];   // [2,E] paper ->author
  const int*   e_cites  = (const int*)d_in[4];   // [2,E] paper ->paper
  const float* Wl       = (const float*)d_in[5]; // [L,3,H,H]
  const float* bl       = (const float*)d_in[6]; // [L,3,H]
  const float* Wr       = (const float*)d_in[7]; // [L,3,H,H]

  const long long NA = in_sizes[0] / HDIM;
  const long long NP = in_sizes[1] / HDIM;
  const long long E  = in_sizes[2] / 2;

  float* out_a = (float*)d_out;                       // [NA*H]
  float* out_p = (float*)d_out + (size_t)NA * HDIM;   // [NP*H]

  // ---- carve workspace ----
  char* ws = (char*)d_ws;
  size_t off = 0;
  auto carve = [&](size_t bytes) -> void* {
    void* p = ws + off;
    off = align256(off + bytes);
    return p;
  };
  float*  xa_buf0 = (float*)carve((size_t)NA * HDIM * 4);
  float*  xa_buf1 = (float*)carve((size_t)NA * HDIM * 4);
  float*  xp_buf0 = (float*)carve((size_t)NP * HDIM * 4);
  float*  xp_buf1 = (float*)carve((size_t)NP * HDIM * 4);
  float*  agg_p   = (float*)carve((size_t)NP * HDIM * 4);
  float*  agg_a   = (float*)carve((size_t)NA * HDIM * 4);
  float*  cnt_p   = (float*)carve((size_t)NP * 4);
  float*  cnt_a   = (float*)carve((size_t)NA * 4);
  bf16_t* mean_p  = (bf16_t*)carve((size_t)NP * HDIM * 2);
  bf16_t* mean_a  = (bf16_t*)carve((size_t)NA * HDIM * 2);
  bf16_t* xbf_p   = (bf16_t*)carve((size_t)NP * HDIM * 2);
  bf16_t* xbf_a   = (bf16_t*)carve((size_t)NA * HDIM * 2);
  bf16_t* WlT     = (bf16_t*)carve((size_t)NLAYER * 3 * HDIM * HDIM * 2);
  bf16_t* WrT     = (bf16_t*)carve((size_t)NLAYER * 3 * HDIM * HDIM * 2);

  auto blocks = [](long long n) -> unsigned { return (unsigned)((n + 255) / 256); };

  // ---- one-time weight transpose + bf16 convert ----
  const long long wtot = (long long)NLAYER * 3 * HDIM * HDIM;
  transpose_cvt_w<<<blocks(wtot), 256, 0, stream>>>(Wl, WlT, wtot);
  transpose_cvt_w<<<blocks(wtot), 256, 0, stream>>>(Wr, WrT, wtot);

  const float* xa_cur = x_author;
  const float* xp_cur = x_paper;

  for (int l = 0; l < NLAYER; ++l) {
    float* xa_nxt = (l == NLAYER - 1) ? out_a : ((l & 1) ? xa_buf1 : xa_buf0);
    float* xp_nxt = (l == NLAYER - 1) ? out_p : ((l & 1) ? xp_buf1 : xp_buf0);

    const size_t mstride = (size_t)HDIM * HDIM;
    const bf16_t* WlT0 = WlT + ((size_t)l * 3 + 0) * mstride;
    const bf16_t* WlT1 = WlT + ((size_t)l * 3 + 1) * mstride;
    const bf16_t* WlT2 = WlT + ((size_t)l * 3 + 2) * mstride;
    const bf16_t* WrT0 = WrT + ((size_t)l * 3 + 0) * mstride;
    const bf16_t* WrT1 = WrT + ((size_t)l * 3 + 1) * mstride;
    const bf16_t* WrT2 = WrT + ((size_t)l * 3 + 2) * mstride;
    const float* bl0 = bl + ((size_t)l * 3 + 0) * HDIM;
    const float* bl1 = bl + ((size_t)l * 3 + 1) * HDIM;
    const float* bl2 = bl + ((size_t)l * 3 + 2) * HDIM;

    // root-term features in bf16
    cvt_f32_bf16<<<blocks(NP * HDIM), 256, 0, stream>>>(xp_cur, xbf_p, NP * HDIM);
    cvt_f32_bf16<<<blocks(NA * HDIM), 256, 0, stream>>>(xa_cur, xbf_a, NA * HDIM);

    const unsigned gP = (unsigned)(NP / 16);   // one block per 16-row M-tile
    const unsigned gA = (unsigned)(NA / 16);

    // --- papers, edge type 0: author -> paper (writes) ---
    zero_f32<<<blocks(NP * HDIM), 256, 0, stream>>>(agg_p, NP * HDIM);
    zero_f32<<<blocks(NP), 256, 0, stream>>>(cnt_p, NP);
    scatter_add<<<(unsigned)E, 256, 0, stream>>>(e_writes, e_writes + E, xa_cur, agg_p, cnt_p);
    mean_cvt<<<blocks(NP * HDIM), 256, 0, stream>>>(agg_p, cnt_p, mean_p, NP * HDIM);
    gemm2_wmma<0><<<gP, 256, 0, stream>>>(mean_p, WlT0, xbf_p, WrT0, bl0, xp_nxt);

    // --- papers, edge type 2: paper -> paper (cites), accumulate ---
    zero_f32<<<blocks(NP * HDIM), 256, 0, stream>>>(agg_p, NP * HDIM);
    zero_f32<<<blocks(NP), 256, 0, stream>>>(cnt_p, NP);
    scatter_add<<<(unsigned)E, 256, 0, stream>>>(e_cites, e_cites + E, xp_cur, agg_p, cnt_p);
    mean_cvt<<<blocks(NP * HDIM), 256, 0, stream>>>(agg_p, cnt_p, mean_p, NP * HDIM);
    gemm2_wmma<1><<<gP, 256, 0, stream>>>(mean_p, WlT2, xbf_p, WrT2, bl2, xp_nxt);

    // --- authors, edge type 1: paper -> author (rev) ---
    zero_f32<<<blocks(NA * HDIM), 256, 0, stream>>>(agg_a, NA * HDIM);
    zero_f32<<<blocks(NA), 256, 0, stream>>>(cnt_a, NA);
    scatter_add<<<(unsigned)E, 256, 0, stream>>>(e_rev, e_rev + E, xp_cur, agg_a, cnt_a);
    mean_cvt<<<blocks(NA * HDIM), 256, 0, stream>>>(agg_a, cnt_a, mean_a, NA * HDIM);
    gemm2_wmma<0><<<gA, 256, 0, stream>>>(mean_a, WlT1, xbf_a, WrT1, bl1, xa_nxt);

    xa_cur = xa_nxt;
    xp_cur = xp_nxt;
  }
}
